// GrassmannMixing_446676598966
// MI455X (gfx1250) — matrix operand order repaired and checked by reading, not compiled, not run on hardware
//
#include <hip/hip_runtime.h>
#include <hip/hip_bf16.h>
#include <math.h>

// ---------------------------------------------------------------------------
// Problem constants (from the reference)
// ---------------------------------------------------------------------------
#define BATCH   4
#define SEQN    4096
#define ROWS    16384      // BATCH * SEQN
#define DMODEL  1024
#define DLOW    32
#define NPAIR   496        // DLOW*(DLOW-1)/2
#define KPAD    512        // NPAIR padded to a multiple of 32
#define KGATE   2048       // concat K for the gate GEMM
#define EPSNRM  1e-6f

// LDS B-tile: [128 columns][40 halves]; 40-half stride => 80B column pitch:
// bank(col) = 20*col mod 64 is a permutation over 16 consecutive cols
// (conflict-free ds_load_b128) and every 16B chunk stays 16B-aligned.
#define BSTRIDE 40

typedef __attribute__((ext_vector_type(16))) _Float16 v16h;
typedef __attribute__((ext_vector_type(8)))  _Float16 v8h;
typedef __attribute__((ext_vector_type(4)))  _Float16 v4h;
typedef __attribute__((ext_vector_type(8)))  float    v8f;
typedef int v4i_gcc __attribute__((vector_size(16)));   // matches builtin param

#define AS1 __attribute__((address_space(1)))
#define AS3 __attribute__((address_space(3)))

// --------------------- CDNA5 async global->LDS path ------------------------
#if defined(__has_builtin)
#  if __has_builtin(__builtin_amdgcn_global_load_async_to_lds_b128)
#    define HAVE_ASYNC_LDS 1
#  endif
#endif
#ifndef HAVE_ASYNC_LDS
#  define HAVE_ASYNC_LDS 0
#endif

#if HAVE_ASYNC_LDS
#  if __has_builtin(__builtin_amdgcn_s_wait_asynccnt)
#    define WAIT_ASYNC(n) __builtin_amdgcn_s_wait_asynccnt(n)
#  else
#    define WAIT_ASYNC(n) asm volatile("s_wait_asynccnt " #n ::: "memory")
#  endif
#else
#  define WAIT_ASYNC(n) do {} while (0)
#endif

// ---------------------------------------------------------------------------
// WMMA fragment helpers. Layouts per CDNA5 ISA 7.12.2 (wave32):
//  A (16x32 f16): lanes 0-15 row M=lane,     K = kb+{0..7, 16..23}
//                 lanes 16-31 row M=lane-16, K = kb+{8..15, 24..31}
//  B (32x16 f16): lanes 0-15 col N=lane,     K = kb+{0..7, 16..23}
//                 lanes 16-31 col N=lane-16, K = kb+{8..15, 24..31}
// ---------------------------------------------------------------------------
__device__ inline v16h load_A_frag_h(const _Float16* __restrict__ A, int lda,
                                     int m0, int kb, int lane) {
  const int row  = m0 + (lane & 15);
  const int koff = (lane & 16) ? 8 : 0;
  const _Float16* base = A + (size_t)row * lda + kb + koff;
  const v8h lo = *(const v8h*)(base);        // K = kb+koff+{0..7}
  const v8h hi = *(const v8h*)(base + 16);   // K = kb+koff+{16..23}
  v16h a;
#pragma unroll
  for (int e = 0; e < 8; ++e) { a[e] = lo[e]; a[e + 8] = hi[e]; }
  return a;
}

// B fragment from the staged LDS tile (column-local index col0 = nt*16)
__device__ inline v16h load_B_frag_lds(const _Float16* __restrict__ Bs,
                                       int col0, int lane) {
  const int col  = col0 + (lane & 15);
  const int koff = (lane & 16) ? 8 : 0;
  const _Float16* base = Bs + col * BSTRIDE + koff;
  const v8h lo = *(const v8h*)(base);
  const v8h hi = *(const v8h*)(base + 16);
  v16h b;
#pragma unroll
  for (int e = 0; e < 8; ++e) { b[e] = lo[e]; b[e + 8] = hi[e]; }
  return b;
}

// Cooperative fill of one 32(K) x 128(N) f16 B tile into LDS.
// 512 16-byte chunks; 256 threads x 2 chunks; 2 async instructions per wave.
__device__ inline void fill_B_tile(const _Float16* __restrict__ Wt, int ldk,
                                   int n0, int kb, _Float16* __restrict__ buf,
                                   int tid) {
#pragma unroll
  for (int r = 0; r < 2; ++r) {
    const int c   = tid + 256 * r;   // chunk id [0, 512)
    const int col = c >> 2;          // [0, 128)
    const int kc  = (c & 3) * 8;     // halves along K
    const _Float16* g = Wt + (size_t)(n0 + col) * ldk + kb + kc;
    _Float16* l = buf + col * BSTRIDE + kc;
#if HAVE_ASYNC_LDS
    __builtin_amdgcn_global_load_async_to_lds_b128(
        (AS1 v4i_gcc*)(uintptr_t)g,
        (AS3 v4i_gcc*)(uint32_t)(uintptr_t)l, 0, 0);
#else
    *(v8h*)l = *(const v8h*)g;       // synchronous fallback
#endif
  }
}

// ---------------------------------------------------------------------------
// Prep kernels: weight transpose+convert, activation convert.
// ---------------------------------------------------------------------------
__global__ __launch_bounds__(256) void cvt_wexp_kernel(
    const float* __restrict__ W, _Float16* __restrict__ Wt) {
  const int idx = blockIdx.x * 256 + threadIdx.x;   // [0, 1024*512)
  const int n = idx >> 9;
  const int k = idx & (KPAD - 1);
  Wt[idx] = (k < NPAIR) ? (_Float16)W[(size_t)k * DMODEL + n] : (_Float16)0.0f;
}

__global__ __launch_bounds__(256) void cvt_wgate_kernel(
    const float* __restrict__ W, _Float16* __restrict__ Wt) {
  const int idx = blockIdx.x * 256 + threadIdx.x;   // [0, 1024*2048)
  const int n = idx >> 11;
  const int k = idx & (KGATE - 1);
  Wt[idx] = (_Float16)W[(size_t)k * DMODEL + n];
}

__global__ __launch_bounds__(256) void cvt_x_kernel(
    const float* __restrict__ x, _Float16* __restrict__ xh) {
  const int q = blockIdx.x * 256 + threadIdx.x;     // quad index
  const float4 v = ((const float4*)x)[q];
  v4h h; h[0] = (_Float16)v.x; h[1] = (_Float16)v.y;
         h[2] = (_Float16)v.z; h[3] = (_Float16)v.w;
  *(v4h*)(xh + (size_t)q * 4) = h;
}

// ---------------------------------------------------------------------------
// Kernel 1: z = x @ W_red + b_red          (ROWS x 1024) @ (1024 x 32)
// ---------------------------------------------------------------------------
__global__ __launch_bounds__(256) void zred_kernel(
    const float* __restrict__ x, const float* __restrict__ Wred,
    const float* __restrict__ bred, float* __restrict__ z) {
  __shared__ float xs[8][DMODEL];
  const int wave = threadIdx.x >> 5;
  const int lane = threadIdx.x & 31;
  const int row  = blockIdx.x * 8 + wave;
  const float* xr = x + (size_t)row * DMODEL;

#pragma unroll
  for (int c = 0; c < 8; ++c) {
    const int idx = (c * 32 + lane) * 4;
    *(float4*)&xs[wave][idx] = *(const float4*)&xr[idx];
  }
  __syncthreads();

  float acc = bred[lane];
#pragma unroll 4
  for (int k = 0; k < DMODEL; ++k)
    acc = fmaf(xs[wave][k], Wred[k * DLOW + lane], acc);

  z[(size_t)row * DLOW + lane] = acc;
}

// ---------------------------------------------------------------------------
// Kernel 2: Plücker minors per lag, L2-normalize, average over lags.
// One wave per row; cross-lane gathers via __shfl; emits f16 padded rows.
// ---------------------------------------------------------------------------
__global__ __launch_bounds__(256) void plucker_kernel(
    const float* __restrict__ z, _Float16* __restrict__ pavg) {
  const int wave = threadIdx.x >> 5;
  const int lane = threadIdx.x & 31;
  const int row  = blockIdx.x * 8 + wave;   // uniform within the wave
  const int t    = row & (SEQN - 1);

  int ii[16], jj[16];
#pragma unroll
  for (int u = 0; u < 16; ++u) {
    const int p = lane * 16 + u;
    int i = 0, rem = p;
    if (p < NPAIR) {
      while (rem >= (DLOW - 1 - i)) { rem -= (DLOW - 1 - i); ++i; }
      ii[u] = i; jj[u] = i + 1 + rem;
    } else { ii[u] = 0; jj[u] = 1; }
  }

  float acc[16];
#pragma unroll
  for (int u = 0; u < 16; ++u) acc[u] = 0.0f;
  float cnt = 0.0f;

  const float zc = z[(size_t)row * DLOW + lane];

  const int lags[4] = {1, 2, 4, 8};
#pragma unroll
  for (int li = 0; li < 4; ++li) {
    const int lag = lags[li];
    if (t < lag) continue;                  // wave-uniform branch
    const float zp = z[(size_t)(row - lag) * DLOW + lane];

    float vals[16];
    float ss = 0.0f;
#pragma unroll
    for (int u = 0; u < 16; ++u) {
      const float zci = __shfl(zc, ii[u], 32);
      const float zcj = __shfl(zc, jj[u], 32);
      const float zpi = __shfl(zp, ii[u], 32);
      const float zpj = __shfl(zp, jj[u], 32);
      float v = zci * zpj - zcj * zpi;
      if (lane * 16 + u >= NPAIR) v = 0.0f;
      vals[u] = v;
      ss = fmaf(v, v, ss);
    }
#pragma unroll
    for (int off = 16; off > 0; off >>= 1)
      ss += __shfl_xor(ss, off, 32);

    const float inv = 1.0f / fmaxf(sqrtf(ss), EPSNRM);
#pragma unroll
    for (int u = 0; u < 16; ++u) acc[u] = fmaf(vals[u], inv, acc[u]);
    cnt += 1.0f;
  }

  const float rc = 1.0f / fmaxf(cnt, 1.0f);
  v16h outv;
#pragma unroll
  for (int u = 0; u < 16; ++u) outv[u] = (_Float16)(acc[u] * rc);
  *(v16h*)(pavg + (size_t)row * KPAD + lane * 16) = outv;   // p>=496 -> 0 pad
}

// ---------------------------------------------------------------------------
// Kernel 3: g_avg = plucker_avg @ W_exp + b_exp -> f16 [ROWS][1024]
// 128x128 block tile, 8 waves x (16M x 128N); B tile double-buffered in LDS
// via async global->LDS; K = 512 (16 tiles). All 8 B fragments are
// materialized before the WMMA burst so the ds_loads batch into one clause.
// ---------------------------------------------------------------------------
__global__ __launch_bounds__(256) void gemm_exp_kernel(
    const _Float16* __restrict__ pavg, const _Float16* __restrict__ Wt,
    const float* __restrict__ bexp, _Float16* __restrict__ gavgh) {
  __shared__ _Float16 Bs[2][128 * BSTRIDE];
  const int tid  = threadIdx.x;
  const int lane = tid & 31;
  const int wave = tid >> 5;
  const int m0   = blockIdx.x * 128 + wave * 16;
  const int n0   = blockIdx.y * 128;

  const v8f zero8 = {0.f, 0.f, 0.f, 0.f, 0.f, 0.f, 0.f, 0.f};
  v8f acc[8];
#pragma unroll
  for (int nt = 0; nt < 8; ++nt) acc[nt] = zero8;

  fill_B_tile(Wt, KPAD, n0, 0, Bs[0], tid);

  const int NIT = KPAD / 32;   // 16
  for (int i = 0; i < NIT; ++i) {
    const int cur = i & 1;
    if (i + 1 < NIT) {
      fill_B_tile(Wt, KPAD, n0, (i + 1) * 32, Bs[cur ^ 1], tid);
      WAIT_ASYNC(2);           // fill(i) complete (in-order), fill(i+1) in flight
    } else {
      WAIT_ASYNC(0);
    }
    __syncthreads();

    const int kb = i * 32;
    __builtin_prefetch(pavg + (size_t)(m0 + (lane & 15)) * KPAD + kb + 32, 0, 3);
    const v16h a = load_A_frag_h(pavg, KPAD, m0, kb, lane);
    v16h bf[8];
#pragma unroll
    for (int nt = 0; nt < 8; ++nt)
      bf[nt] = load_B_frag_lds(Bs[cur], nt * 16, lane);
#pragma unroll
    for (int nt = 0; nt < 8; ++nt)
      acc[nt] = __builtin_amdgcn_wmma_f32_16x16x32_f16(
          false, a, false, bf[nt], (short)0, acc[nt], false, false);
    __syncthreads();           // all waves done reading Bs[cur] before refill
  }

  const int mbase = m0 + ((lane & 16) ? 8 : 0);
#pragma unroll
  for (int nt = 0; nt < 8; ++nt) {
    const int col  = n0 + nt * 16 + (lane & 15);
    const float bi = bexp[col];
#pragma unroll
    for (int vr = 0; vr < 8; ++vr)
      gavgh[(size_t)(mbase + vr) * DMODEL + col] = (_Float16)(acc[nt][vr] + bi);
  }
}

// ---------------------------------------------------------------------------
// Kernel 4: gate GEMM + fused epilogue. Single 64-tile K pipeline:
// tiles 0..31 use A=xh, tiles 32..63 use A=gavgh; B = Wt_gate (K=2048).
// out = sigmoid(acc + b_gate) * x + (1 - sigmoid) * g_avg
// ---------------------------------------------------------------------------
__global__ __launch_bounds__(256) void gemm_gate_kernel(
    const float* __restrict__ x, const _Float16* __restrict__ xh,
    const _Float16* __restrict__ gavgh, const _Float16* __restrict__ Wt,
    const float* __restrict__ bgate, float* __restrict__ out) {
  __shared__ _Float16 Bs[2][128 * BSTRIDE];
  const int tid  = threadIdx.x;
  const int lane = tid & 31;
  const int wave = tid >> 5;
  const int m0   = blockIdx.x * 128 + wave * 16;
  const int n0   = blockIdx.y * 128;

  const v8f zero8 = {0.f, 0.f, 0.f, 0.f, 0.f, 0.f, 0.f, 0.f};
  v8f acc[8];
#pragma unroll
  for (int nt = 0; nt < 8; ++nt) acc[nt] = zero8;

  fill_B_tile(Wt, KGATE, n0, 0, Bs[0], tid);

  const int NIT = KGATE / 32;  // 64
  for (int i = 0; i < NIT; ++i) {
    const int cur = i & 1;
    if (i + 1 < NIT) {
      fill_B_tile(Wt, KGATE, n0, (i + 1) * 32, Bs[cur ^ 1], tid);
      WAIT_ASYNC(2);
    } else {
      WAIT_ASYNC(0);
    }
    __syncthreads();

    const int kb  = i * 32;
    const _Float16* Asrc = (i < NIT / 2) ? xh : gavgh;   // wave-uniform
    const int akb = (i < NIT / 2) ? kb : kb - DMODEL;
    __builtin_prefetch(Asrc + (size_t)(m0 + (lane & 15)) * DMODEL + akb + 32, 0, 3);
    const v16h a = load_A_frag_h(Asrc, DMODEL, m0, akb, lane);
    v16h bf[8];
#pragma unroll
    for (int nt = 0; nt < 8; ++nt)
      bf[nt] = load_B_frag_lds(Bs[cur], nt * 16, lane);
#pragma unroll
    for (int nt = 0; nt < 8; ++nt)
      acc[nt] = __builtin_amdgcn_wmma_f32_16x16x32_f16(
          false, a, false, bf[nt], (short)0, acc[nt], false, false);
    __syncthreads();
  }

  const int mbase = m0 + ((lane & 16) ? 8 : 0);
#pragma unroll
  for (int nt = 0; nt < 8; ++nt) {
    const int col  = n0 + nt * 16 + (lane & 15);
    const float bi = bgate[col];
#pragma unroll
    for (int vr = 0; vr < 8; ++vr) {
      const int m    = mbase + vr;
      const float tv = acc[nt][vr] + bi;
      const float g  = 1.0f / (1.0f + __expf(-tv));
      const float xv = x[(size_t)m * DMODEL + col];
      const float gv = (float)gavgh[(size_t)m * DMODEL + col];
      out[(size_t)m * DMODEL + col] = g * xv + (1.0f - g) * gv;
    }
  }
}

// ---------------------------------------------------------------------------
// Launch. Inputs: 0:x 1:W_red 2:b_red 3:W_exp 4:b_exp 5:W_gate 6:b_gate
// Workspace: z f32 2MB | pavg f16 16MB | xh f16 32MB | gavgh f16 32MB |
//            WtExp f16 1MB | WtGate f16 4MB   (total 87 MB)
// ---------------------------------------------------------------------------
extern "C" void kernel_launch(void* const* d_in, const int* in_sizes, int n_in,
                              void* d_out, int out_size, void* d_ws, size_t ws_size,
                              hipStream_t stream) {
  const float* x      = (const float*)d_in[0];
  const float* W_red  = (const float*)d_in[1];
  const float* b_red  = (const float*)d_in[2];
  const float* W_exp  = (const float*)d_in[3];
  const float* b_exp  = (const float*)d_in[4];
  const float* W_gate = (const float*)d_in[5];
  const float* b_gate = (const float*)d_in[6];
  float* out = (float*)d_out;

  float*    z      = (float*)d_ws;
  _Float16* pavg   = (_Float16*)(z + (size_t)ROWS * DLOW);
  _Float16* xh     = pavg  + (size_t)ROWS * KPAD;
  _Float16* gavgh  = xh    + (size_t)ROWS * DMODEL;
  _Float16* WtExp  = gavgh + (size_t)ROWS * DMODEL;
  _Float16* WtGate = WtExp + (size_t)DMODEL * KPAD;

  cvt_wexp_kernel <<<(DMODEL * KPAD) / 256, 256, 0, stream>>>(W_exp, WtExp);
  cvt_wgate_kernel<<<(DMODEL * KGATE) / 256, 256, 0, stream>>>(W_gate, WtGate);
  cvt_x_kernel    <<<((size_t)ROWS * DMODEL / 4) / 256, 256, 0, stream>>>(x, xh);

  zred_kernel   <<<ROWS / 8, 256, 0, stream>>>(x, W_red, b_red, z);
  plucker_kernel<<<ROWS / 8, 256, 0, stream>>>(z, pavg);

  gemm_exp_kernel <<<dim3(ROWS / 128, DMODEL / 128), 256, 0, stream>>>(
      pavg, WtExp, b_exp, gavgh);
  gemm_gate_kernel<<<dim3(ROWS / 128, DMODEL / 128), 256, 0, stream>>>(
      x, xh, gavgh, WtGate, b_gate, out);
}
